// BCQConv1D_87814901334607
// MI455X (gfx1250) — matrix operand order repaired and checked by reading, not compiled, not run on hardware
//
#include <hip/hip_runtime.h>

typedef __attribute__((ext_vector_type(16))) _Float16 v16h;
typedef __attribute__((ext_vector_type(8)))  _Float16 v8h;
typedef __attribute__((ext_vector_type(8)))  float    v8f;
typedef __attribute__((ext_vector_type(4)))  float    v4f;

// Problem dims (fixed by reference: B=4, S=2048, I=4096, O=4096, G=32, A=128, QB=3)
#define M_DIM 8192
#define K_DIM 4096
#define N_DIM 4096
#define G_DIM 32
#define A_DIM 128
#define QB    3

// GEMM tiling
#define BM   128
#define BN   128
#define BK   64          // K per pipeline stage (two 16x16x32 WMMA K-steps)
#define LDT  72          // padded LDS row stride in halves (144B = 9*16B: aligned + conflict-free)
#define ABUF (BM * LDT)  // halves per LDS buffer (A and B each, double buffered)

// ---------------------------------------------------------------------------
// Phase 0a: x f32 -> f16 (one pass; makes GEMM A-tiles pure async byte copies)
// ---------------------------------------------------------------------------
__global__ __launch_bounds__(256) void cvt_x_f16(const float* __restrict__ x,
                                                 _Float16* __restrict__ xh) {
  size_t idx = ((size_t)blockIdx.x * 256 + threadIdx.x) * 8;  // M*K / 8 threads
  v4f a = *(const v4f*)(x + idx);
  v4f b = *(const v4f*)(x + idx + 4);
  v8h h;
#pragma unroll
  for (int j = 0; j < 4; ++j) { h[j] = (_Float16)a[j]; h[j + 4] = (_Float16)b[j]; }
  *(v8h*)(xh + idx) = h;
}

// ---------------------------------------------------------------------------
// Phase 0b: reconstruct BCQ weight  W[o,i] = sum_b alpha[o,g,b]*binary[o,g,a,b]
// ---------------------------------------------------------------------------
__global__ __launch_bounds__(256) void bcq_build_weight(
    const float* __restrict__ alpha,   // [O, G, QB]
    const float* __restrict__ binary,  // [O, G, A, QB]
    _Float16* __restrict__ wh)         // [O, K] f16
{
  int idx = blockIdx.x * blockDim.x + threadIdx.x;  // over O*K
  if (idx >= N_DIM * K_DIM) return;
  int o = idx >> 12;
  int i = idx & 4095;
  int g = i >> 7;
  int a = i & 127;
  const float* al = alpha  + (size_t)(o * G_DIM + g) * QB;
  const float* bi = binary + ((size_t)(o * G_DIM + g) * A_DIM + a) * QB;
  float w = al[0] * bi[0] + al[1] * bi[1] + al[2] * bi[2];
  wh[idx] = (_Float16)w;
}

static __device__ __forceinline__ v16h cat8(v8h a, v8h b) {
  v16h r;
#pragma unroll
  for (int i = 0; i < 8; ++i) { r[i] = a[i]; r[i + 8] = b[i]; }
  return r;
}

// 4 async 16B memory->LDS copies; per ISA the instruction offset is added to
// BOTH the LDS and global address, so one (lds, gaddr) pair covers 64B.
static __device__ __forceinline__ void async_cp64B(unsigned lds, const _Float16* g) {
  asm volatile(
      "global_load_async_to_lds_b128 %0, %1, off\n\t"
      "global_load_async_to_lds_b128 %0, %1, off offset:16\n\t"
      "global_load_async_to_lds_b128 %0, %1, off offset:32\n\t"
      "global_load_async_to_lds_b128 %0, %1, off offset:48"
      :: "v"(lds), "v"(g) : "memory");
}

static __device__ __forceinline__ void wait_async0() {
  asm volatile("s_wait_asynccnt 0x0" ::: "memory");
}

// ---------------------------------------------------------------------------
// Phase 1: out[M,N] = xh[M,K] * wh[N,K]^T + bias.  f16 WMMA, f32 accumulate.
// 256 threads = 8 waves (wave32); block tile 128x128, wave tile 64x32.
// Double-buffered LDS fed by ASYNCcnt-tracked async-to-LDS copies.
// ---------------------------------------------------------------------------
__global__ __launch_bounds__(256) void bcq_gemm_wmma(
    const _Float16* __restrict__ xh,    // [M, K] f16
    const _Float16* __restrict__ wh,    // [N, K] f16
    const float*    __restrict__ bias,  // [N]
    float*          __restrict__ out)   // [M, N]
{
  __shared__ _Float16 Ash[2 * ABUF];
  __shared__ _Float16 Bsh[2 * ABUF];

  const int tid    = threadIdx.x;
  const int wave   = tid >> 5;
  const int lane   = tid & 31;
  const int wave_m = wave >> 2;   // 0..1
  const int wave_n = wave & 3;    // 0..3
  const int lhalf  = lane >> 4;   // 0/1
  const int l16    = lane & 15;

  const int blk_n = blockIdx.x;
  const int blk_m = blockIdx.y;

  // Staging map: thread t copies row r = t/2, 64B chunk at half-offset (t&1)*32.
  const int srow = tid >> 1;
  const int scol = (tid & 1) << 5;  // 0 or 32 halves

  const _Float16* gA = xh + (size_t)(blk_m * BM + srow) * K_DIM + scol;
  const _Float16* gB = wh + (size_t)(blk_n * BN + srow) * K_DIM + scol;

  const unsigned lA0 = (unsigned)(uintptr_t)&Ash[srow * LDT + scol];
  const unsigned lA1 = (unsigned)(uintptr_t)&Ash[ABUF + srow * LDT + scol];
  const unsigned lB0 = (unsigned)(uintptr_t)&Bsh[srow * LDT + scol];
  const unsigned lB1 = (unsigned)(uintptr_t)&Bsh[ABUF + srow * LDT + scol];

  v8f acc[4][2];
#pragma unroll
  for (int tm = 0; tm < 4; ++tm)
#pragma unroll
    for (int tn = 0; tn < 2; ++tn)
#pragma unroll
      for (int r = 0; r < 8; ++r) acc[tm][tn][r] = 0.0f;

  // Prologue: stage 0 into buffer 0.
  async_cp64B(lA0, gA);
  async_cp64B(lB0, gB);

  const int NSTEP = K_DIM / BK;  // 64
  for (int i = 0; i < NSTEP; ++i) {
    wait_async0();     // my stage-i copies have landed in LDS
    __syncthreads();   // everyone's copies visible; next buffer free

    if (i + 1 < NSTEP) {  // overlap stage i+1 copies with stage i math
      const _Float16* nA = gA + (size_t)(i + 1) * BK;
      const _Float16* nB = gB + (size_t)(i + 1) * BK;
      const unsigned la = ((i + 1) & 1) ? lA1 : lA0;
      const unsigned lb = ((i + 1) & 1) ? lB1 : lB0;
      async_cp64B(la, nA);
      async_cp64B(lb, nB);
    }

    const _Float16* Ab = Ash + (i & 1) * ABUF;
    const _Float16* Bb = Bsh + (i & 1) * ABUF;

#pragma unroll
    for (int ks = 0; ks < BK; ks += 32) {
      v16h afrag[4], bfrag[2];
      const int koffA = ks + (lhalf << 3);  // lanes 0-15: K 0..7,16..23 ; 16-31: 8..15,24..31
#pragma unroll
      for (int tm = 0; tm < 4; ++tm) {
        const int m = wave_m * 64 + tm * 16 + l16;
        v8h c0 = *(const v8h*)&Ab[m * LDT + koffA];
        v8h c1 = *(const v8h*)&Ab[m * LDT + koffA + 16];
        afrag[tm] = cat8(c0, c1);
      }
      const int koffB = ks + (lhalf << 3);  // lanes 0-15: K 0..7,16..23 ; 16-31: 8..15,24..31
#pragma unroll
      for (int tn = 0; tn < 2; ++tn) {
        const int n = wave_n * 32 + tn * 16 + l16;
        v8h c0 = *(const v8h*)&Bb[n * LDT + koffB];
        v8h c1 = *(const v8h*)&Bb[n * LDT + koffB + 16];
        bfrag[tn] = cat8(c0, c1);
      }
#pragma unroll
      for (int tm = 0; tm < 4; ++tm)
#pragma unroll
        for (int tn = 0; tn < 2; ++tn)
          acc[tm][tn] = __builtin_amdgcn_wmma_f32_16x16x32_f16(
              false, afrag[tm], false, bfrag[tn],
              (short)0, acc[tm][tn], false, false);
    }
  }

  // Epilogue: + bias, coalesced f32 stores (lane%16 spans consecutive n).
  const int rowbase = blk_m * BM + wave_m * 64;
  const int colbase = blk_n * BN + wave_n * 32;
  float bv[2];
#pragma unroll
  for (int tn = 0; tn < 2; ++tn) bv[tn] = bias[colbase + tn * 16 + l16];

#pragma unroll
  for (int tm = 0; tm < 4; ++tm)
#pragma unroll
    for (int tn = 0; tn < 2; ++tn) {
      const int n = colbase + tn * 16 + l16;
#pragma unroll
      for (int r = 0; r < 8; ++r) {
        const int m = rowbase + tm * 16 + lhalf * 8 + r;
        out[(size_t)m * N_DIM + n] = acc[tm][tn][r] + bv[tn];
      }
    }
}

extern "C" void kernel_launch(void* const* d_in, const int* in_sizes, int n_in,
                              void* d_out, int out_size, void* d_ws, size_t ws_size,
                              hipStream_t stream) {
  (void)in_sizes; (void)n_in; (void)out_size; (void)ws_size;
  const float* x      = (const float*)d_in[0];  // [4,2048,4096]
  const float* alpha  = (const float*)d_in[1];  // [4096,32,3]
  const float* bias   = (const float*)d_in[2];  // [4096]
  const float* binary = (const float*)d_in[3];  // [4096,32,128,3]
  float* out = (float*)d_out;                   // [4,2048,4096]

  // Workspace layout: xh f16 [M,K] (64 MiB) then wh f16 [N,K] (32 MiB).
  _Float16* xh = (_Float16*)d_ws;
  _Float16* wh = xh + (size_t)M_DIM * K_DIM;

  // Phase 0: one memory-bound pass each (~8 + ~10 us at 23.3 TB/s).
  cvt_x_f16<<<(M_DIM * (size_t)K_DIM) / (256 * 8), 256, 0, stream>>>(x, xh);
  bcq_build_weight<<<(N_DIM * K_DIM) / 256, 256, 0, stream>>>(alpha, binary, wh);

  // Phase 1: WMMA GEMM, 32 x 64 blocks of 128x128.
  dim3 grid(N_DIM / BN, M_DIM / BM);
  bcq_gemm_wmma<<<grid, 256, 0, stream>>>(xh, wh, bias, out);
}
